// mHCTransformerBlock_43258910605337
// MI455X (gfx1250) — compile-verified, matrix operand order checked by reading, not run on hardware
//
#include <hip/hip_runtime.h>
#include <cstddef>
#include <cstdint>

#define DEVI __device__ __forceinline__

typedef __attribute__((ext_vector_type(8)))  float  f32x8;
typedef __attribute__((ext_vector_type(8)))  __bf16 bf16x8;
typedef __attribute__((ext_vector_type(16))) __bf16 bf16x16;
typedef __attribute__((ext_vector_type(4)))  int    i32x4;

static constexpr int BB = 2, LL = 2048, CC = 1024, HH = 16, FF = 2752;
static constexpr int TT = BB * LL;   // 4096 tokens
static constexpr int NC = 4 * CC;    // 4096
static constexpr int FPAD = 2816;    // FF padded to multiple of 128

// ---------------- helpers ----------------
DEVI unsigned short f2bf(float f) {
  unsigned int u = __float_as_uint(f);
  u += 0x7FFFu + ((u >> 16) & 1u);      // round-to-nearest-even
  return (unsigned short)(u >> 16);
}

DEVI float warp_sum(float v) {
  for (int off = 16; off > 0; off >>= 1) v += __shfl_xor(v, off, 32);
  return v;
}

DEVI bf16x16 make_frag(const unsigned short* p0, const unsigned short* p1) {
  bf16x8 lo = *(const bf16x8*)p0;
  bf16x8 hi = *(const bf16x8*)p1;
  return __builtin_shufflevector(lo, hi, 0,1,2,3,4,5,6,7,8,9,10,11,12,13,14,15);
}

DEVI f32x8 wmma_bf16(bf16x16 a, bf16x16 b, f32x8 c) {
  return __builtin_amdgcn_wmma_f32_16x16x32_bf16(false, a, false, b, (short)0, c, false, false);
}

// ---- CDNA5 async global->LDS copy (ASYNCcnt path), with plain fallback ----
#if __has_builtin(__builtin_amdgcn_global_load_async_to_lds_b128)
#define USE_ASYNC_LDS 1
DEVI void async_cp16(unsigned short* lds, const unsigned short* g) {
  __builtin_amdgcn_global_load_async_to_lds_b128(
      (__attribute__((address_space(1))) i32x4*)(g),
      (__attribute__((address_space(3))) i32x4*)(lds), 0, 0);
}
DEVI void async_wait0() { asm volatile("s_wait_asynccnt 0x0" ::: "memory"); }
#else
#define USE_ASYNC_LDS 0
DEVI void async_wait0() {}
#endif

// stage one 128x32 bf16 tile pair (A and B) into LDS (40-elem padded rows)
DEVI void stage_tile(unsigned short* Asb, unsigned short* Bsb,
                     const unsigned short* aSrc, const unsigned short* bSrc,
                     int lrow, int lcol) {
#if USE_ASYNC_LDS
  async_cp16(Asb + lrow * 40 + lcol,     aSrc);
  async_cp16(Asb + lrow * 40 + lcol + 8, aSrc + 8);
  async_cp16(Bsb + lrow * 40 + lcol,     bSrc);
  async_cp16(Bsb + lrow * 40 + lcol + 8, bSrc + 8);
#else
  const uint4* a = (const uint4*)aSrc;
  const uint4* b = (const uint4*)bSrc;
  uint4 a0 = a[0], a1 = a[1], b0 = b[0], b1 = b[1];
  *(uint4*)(Asb + lrow * 40 + lcol)     = a0;
  *(uint4*)(Asb + lrow * 40 + lcol + 8) = a1;
  *(uint4*)(Bsb + lrow * 40 + lcol)     = b0;
  *(uint4*)(Bsb + lrow * 40 + lcol + 8) = b1;
#endif
}

// ---------------- f32 -> bf16 conversion with zero-padding ----------------
__global__ void cvt_bf16_kernel(const float* __restrict__ in,
                                unsigned short* __restrict__ out,
                                int n, int ntot) {
  int i = blockIdx.x * 256 + threadIdx.x;
  if (i < ntot) out[i] = (i < n) ? f2bf(in[i]) : (unsigned short)0;
}

// ---------------- mHC stage: rms_apply, phi projections, sinkhorn, h, rmsnorm
__global__ __launch_bounds__(256) void mhc_stage_kernel(
    const float* __restrict__ x,          // [T, 4096]
    const float* __restrict__ phi_pre,    // [4096, 4]
    const float* __restrict__ phi_post,   // [4096, 4]
    const float* __restrict__ phi_res,    // [4096, 16]
    const float* __restrict__ a_pre, const float* __restrict__ a_post,
    const float* __restrict__ a_res,
    const float* __restrict__ b_pre, const float* __restrict__ b_post,
    const float* __restrict__ b_res,
    const float* __restrict__ ln_w,       // [1024]
    float* __restrict__ preo,             // [T, 4]
    float* __restrict__ posto,            // [T, 4]
    float* __restrict__ reso,             // [T, 16]
    unsigned short* __restrict__ hnb)     // [T, 1024] bf16
{
  const int t = blockIdx.x, tid = threadIdx.x;
  const int lane = tid & 31, wv = tid >> 5;
  const float* xr = x + (size_t)t * NC;

  __shared__ float red8[8];
  __shared__ float partial[24 * 8];
  __shared__ float dots[24];
  __shared__ float prev[4];

  // ---- sum of squares over 4096 ----
  float ss = 0.f;
  for (int i = tid; i < NC; i += 256) { float v = xr[i]; ss += v * v; }
  ss = warp_sum(ss);
  if (lane == 0) red8[wv] = ss;
  __syncthreads();
  float ssq = red8[0]+red8[1]+red8[2]+red8[3]+red8[4]+red8[5]+red8[6]+red8[7];
  float rinv = rsqrtf(ssq / (float)NC + 1e-6f);
  __syncthreads();

  // ---- 24 projections of clipped xn ----
  float acc[24];
#pragma unroll
  for (int a = 0; a < 24; a++) acc[a] = 0.f;
  for (int i = tid; i < NC; i += 256) {
    float xv = xr[i] * rinv;
    xv = fminf(fmaxf(xv, -10.f), 10.f);
    const float* pp = phi_pre  + (size_t)i * 4;
    const float* pq = phi_post + (size_t)i * 4;
    const float* pr = phi_res  + (size_t)i * 16;
#pragma unroll
    for (int j = 0; j < 4; j++)  acc[j]     += xv * pp[j];
#pragma unroll
    for (int j = 0; j < 4; j++)  acc[4 + j] += xv * pq[j];
#pragma unroll
    for (int j = 0; j < 16; j++) acc[8 + j] += xv * pr[j];
  }
#pragma unroll
  for (int a = 0; a < 24; a++) {
    float v = warp_sum(acc[a]);
    if (lane == 0) partial[a * 8 + wv] = v;
  }
  __syncthreads();
  if (tid < 24) {
    float s = 0.f;
#pragma unroll
    for (int w = 0; w < 8; w++) s += partial[tid * 8 + w];
    dots[tid] = s;
  }
  __syncthreads();

  // ---- sigmoid maps + sinkhorn (tiny 4x4, serial on thread 0) ----
  if (tid == 0) {
    float ap = a_pre[0], aq = a_post[0], ar = a_res[0];
#pragma unroll
    for (int j = 0; j < 4; j++) {
      float pv = 1.f / (1.f + __expf(-(ap * dots[j] + b_pre[j])));
      prev[j] = pv;
      preo[(size_t)t * 4 + j] = pv;
      posto[(size_t)t * 4 + j] = 2.f / (1.f + __expf(-(aq * dots[4 + j] + b_post[j])));
    }
    float Mp[16];
#pragma unroll
    for (int e = 0; e < 16; e++) {
      float m = ar * dots[8 + e] + b_res[e];
      m = fminf(fmaxf(m, -20.f), 20.f);
      Mp[e] = __expf(m);
    }
    for (int it = 0; it < 20; ++it) {
#pragma unroll
      for (int n = 0; n < 4; n++) {
        float s = Mp[n*4] + Mp[n*4+1] + Mp[n*4+2] + Mp[n*4+3] + 1e-6f;
        float iv = 1.f / s;
#pragma unroll
        for (int m2 = 0; m2 < 4; m2++) Mp[n*4+m2] *= iv;
      }
#pragma unroll
      for (int m2 = 0; m2 < 4; m2++) {
        float s = Mp[m2] + Mp[4+m2] + Mp[8+m2] + Mp[12+m2] + 1e-6f;
        float iv = 1.f / s;
#pragma unroll
        for (int n = 0; n < 4; n++) Mp[n*4+m2] *= iv;
      }
    }
#pragma unroll
    for (int e = 0; e < 16; e++) reso[(size_t)t * 16 + e] = Mp[e];
  }
  __syncthreads();

  // ---- h = pre . x ; rmsnorm(h) * ln_w -> bf16 ----
  float hv[4], hs = 0.f;
#pragma unroll
  for (int ci = 0; ci < 4; ci++) {
    int c = tid + ci * 256;
    float h = prev[0]*xr[c] + prev[1]*xr[CC + c] + prev[2]*xr[2*CC + c] + prev[3]*xr[3*CC + c];
    hv[ci] = h; hs += h * h;
  }
  hs = warp_sum(hs);
  if (lane == 0) red8[wv] = hs;
  __syncthreads();
  float htot = red8[0]+red8[1]+red8[2]+red8[3]+red8[4]+red8[5]+red8[6]+red8[7];
  float hri = rsqrtf(htot / (float)CC + 1e-5f);
#pragma unroll
  for (int ci = 0; ci < 4; ci++) {
    int c = tid + ci * 256;
    hnb[(size_t)t * CC + c] = f2bf(hv[ci] * hri * ln_w[c]);
  }
}

// ---------------- WMMA bf16 GEMM: C[M,N] = A[M,K] * W[Npad,K]^T ----------------
// B rows are zero-padded to the 128-tile boundary, so no load guards in the loop.
// Double-buffered LDS tiles; next tile is prefetched (async-to-LDS when available)
// while the current one feeds the WMMAs.
__global__ __launch_bounds__(256) void gemm_bf16_kernel(
    const unsigned short* __restrict__ Ag,   // [M,K] bf16, K % 32 == 0
    const unsigned short* __restrict__ Bg,   // [Npad,K] bf16 (row = out feature)
    float* __restrict__ Cg,                  // [M,N] f32
    int M, int N, int K)
{
  __shared__ unsigned short As[2][128 * 40];
  __shared__ unsigned short Bs[2][128 * 40];
  const int tid = threadIdx.x;
  const int lane = tid & 31, wave = tid >> 5;
  const int wr = wave & 3, wc = wave >> 2;       // 4 x 2 wave grid
  const int mW = wr * 32, nW = wc * 64;          // wave tile 32x64
  const int mB = blockIdx.y * 128, nB = blockIdx.x * 128;
  const int lrow = tid >> 1, lcol = (tid & 1) * 16;
  const int hm = lane & 15;
  const int c0 = (lane < 16) ? 0 : 8;

  f32x8 acc[2][4];
  f32x8 z = {};
#pragma unroll
  for (int i = 0; i < 2; i++)
#pragma unroll
    for (int j = 0; j < 4; j++) acc[i][j] = z;

  const unsigned short* aBase = Ag + (size_t)(mB + lrow) * K + lcol;
  const unsigned short* bBase = Bg + (size_t)(nB + lrow) * K + lcol;

  stage_tile(As[0], Bs[0], aBase, bBase, lrow, lcol);

  const int nsteps = K / 32;
  for (int it = 0; it < nsteps; ++it) {
    const int cur = it & 1;
    async_wait0();              // this wave's async copies into buffer `cur` done
    __syncthreads();            // everyone's copies done; prev compute retired
    if (it + 1 < nsteps)
      stage_tile(As[cur ^ 1], Bs[cur ^ 1],
                 aBase + (size_t)(it + 1) * 32, bBase + (size_t)(it + 1) * 32,
                 lrow, lcol);

    const unsigned short* Ac = As[cur];
    const unsigned short* Bc = Bs[cur];
    bf16x16 af[2], bfr[4];
#pragma unroll
    for (int i = 0; i < 2; i++) {
      int r = mW + i * 16 + hm;
      af[i] = make_frag(Ac + r * 40 + c0, Ac + r * 40 + c0 + 16);
    }
#pragma unroll
    for (int j = 0; j < 4; j++) {
      int r = nW + j * 16 + hm;
      bfr[j] = make_frag(Bc + r * 40 + c0, Bc + r * 40 + c0 + 16);
    }
#pragma unroll
    for (int i = 0; i < 2; i++)
#pragma unroll
      for (int j = 0; j < 4; j++)
        acc[i][j] = wmma_bf16(af[i], bfr[j], acc[i][j]);
  }

  const int rowoff = (lane >> 4) * 8;
#pragma unroll
  for (int i = 0; i < 2; i++)
#pragma unroll
    for (int j = 0; j < 4; j++)
#pragma unroll
      for (int r = 0; r < 8; r++) {
        int m = mB + mW + i * 16 + rowoff + r;
        int n = nB + nW + j * 16 + hm;
        if (n < N) Cg[(size_t)m * N + n] = acc[i][j][r];
      }
}

// ---------------- RoPE + reformat (Q,K -> [bh,l,dk] bf16; V -> [bh,dk,l] bf16)
__global__ void rope_reformat_kernel(
    const float* __restrict__ qf, const float* __restrict__ kf,
    const float* __restrict__ vf, const int* __restrict__ pos,
    unsigned short* __restrict__ qb, unsigned short* __restrict__ kb,
    unsigned short* __restrict__ vtb)
{
  int idx = blockIdx.x * 256 + threadIdx.x;        // T*H threads
  if (idx >= TT * HH) return;
  int t = idx >> 4, h = idx & 15;
  int b = t >> 11, l = t & (LL - 1);
  float p = (float)pos[l];
  const float* q = qf + (size_t)t * CC + h * 64;
  const float* k = kf + (size_t)t * CC + h * 64;
  const float* v = vf + (size_t)t * CC + h * 64;
  size_t bh = (size_t)(b * HH + h);
  unsigned short* qo = qb + (bh * LL + l) * 64;
  unsigned short* ko = kb + (bh * LL + l) * 64;
  const float lt = logf(10000.0f);
#pragma unroll 4
  for (int i = 0; i < 32; i++) {
    float fr = __expf(-((float)(2 * i) / 64.0f) * lt);
    float sn, cs;
    __sincosf(p * fr, &sn, &cs);
    float qe = q[2*i], qd = q[2*i+1];
    qo[2*i]   = f2bf(qe * cs - qd * sn);
    qo[2*i+1] = f2bf(qe * sn + qd * cs);
    float ke = k[2*i], kd = k[2*i+1];
    ko[2*i]   = f2bf(ke * cs - kd * sn);
    ko[2*i+1] = f2bf(ke * sn + kd * cs);
  }
#pragma unroll 4
  for (int d = 0; d < 64; d++)
    vtb[(bh * 64 + d) * LL + l] = f2bf(v[d]);
}

// ---------------- flash attention: one wave per 16-query tile ----------------
__global__ __launch_bounds__(128) void flash_attn_kernel(
    const unsigned short* __restrict__ qb,   // [B*H, L, 64]
    const unsigned short* __restrict__ kb,   // [B*H, L, 64]
    const unsigned short* __restrict__ vtb,  // [B*H, 64, L]
    unsigned short* __restrict__ aob)        // [T, 1024] bf16 (b,l,h,dk)
{
  __shared__ unsigned short Pl[4][16 * 40];
  const int tid = threadIdx.x, lane = tid & 31, wave = tid >> 5;
  const int bh = blockIdx.x;
  const int b = bh >> 4, h = bh & 15;
  const int q0 = (blockIdx.y * 4 + wave) * 16;
  const int hm = lane & 15, half = lane >> 4;
  const int c0 = half * 8;

  const unsigned short* qrow = qb + ((size_t)bh * LL + q0 + hm) * 64;
  bf16x16 qf0 = make_frag(qrow + c0,      qrow + c0 + 16);
  bf16x16 qf1 = make_frag(qrow + 32 + c0, qrow + 32 + c0 + 16);

  f32x8 z = {};
  f32x8 o[4] = {z, z, z, z};
  float m8[8], l8[8];
#pragma unroll
  for (int r = 0; r < 8; r++) { m8[r] = -1e30f; l8[r] = 0.f; }

  unsigned short* P = &Pl[wave][0];

  for (int k0 = 0; k0 <= q0 + 15; k0 += 32) {
    f32x8 s[2];
#pragma unroll
    for (int sub = 0; sub < 2; sub++) {
      const unsigned short* krow = kb + ((size_t)bh * LL + k0 + sub * 16 + hm) * 64;
      bf16x16 kf0 = make_frag(krow + c0,      krow + c0 + 16);
      bf16x16 kf1 = make_frag(krow + 32 + c0, krow + 32 + c0 + 16);
      f32x8 sz = {};
      sz = wmma_bf16(qf0, kf0, sz);
      sz = wmma_bf16(qf1, kf1, sz);
      s[sub] = sz;
    }
    // scale + causal mask
#pragma unroll
    for (int sub = 0; sub < 2; sub++) {
      int col = k0 + sub * 16 + hm;
#pragma unroll
      for (int r = 0; r < 8; r++) {
        int row = q0 + r + half * 8;
        float v = s[sub][r] * 0.125f;     // 1/sqrt(64)
        s[sub][r] = (col <= row) ? v : -1e30f;
      }
    }
    // online softmax, P -> LDS (bf16)
#pragma unroll
    for (int r = 0; r < 8; r++) {
      float mx = fmaxf(s[0][r], s[1][r]);
      for (int off = 8; off > 0; off >>= 1) mx = fmaxf(mx, __shfl_xor(mx, off, 32));
      float nm = fmaxf(m8[r], mx);
      float sc = __expf(m8[r] - nm);
      m8[r] = nm;
      float p0 = __expf(s[0][r] - nm);
      float p1 = __expf(s[1][r] - nm);
      float rs = p0 + p1;
      for (int off = 8; off > 0; off >>= 1) rs += __shfl_xor(rs, off, 32);
      l8[r] = l8[r] * sc + rs;
#pragma unroll
      for (int j = 0; j < 4; j++) o[j][r] *= sc;
      int prow = r + half * 8;
      P[prow * 40 + hm]      = f2bf(p0);
      P[prow * 40 + 16 + hm] = f2bf(p1);
    }
    // D-layout -> A-layout through LDS (wave-internal, DS ops stay in order)
    bf16x16 pf = make_frag(P + hm * 40 + c0, P + hm * 40 + c0 + 16);
#pragma unroll
    for (int j = 0; j < 4; j++) {
      const unsigned short* vrow = vtb + ((size_t)bh * 64 + j * 16 + hm) * LL + k0;
      bf16x16 vfr = make_frag(vrow + c0, vrow + c0 + 16);
      o[j] = wmma_bf16(pf, vfr, o[j]);
    }
  }

#pragma unroll
  for (int r = 0; r < 8; r++) {
    float inv = 1.f / l8[r];
    int l = q0 + r + half * 8;
    size_t base = ((size_t)(b * LL + l)) * CC + h * 64;
#pragma unroll
    for (int j = 0; j < 4; j++)
      aob[base + j * 16 + hm] = f2bf(o[j][r] * inv);
  }
}

// ---------------- silu(a)*g -> bf16 ----------------
__global__ void silu_mul_kernel(const float* __restrict__ a,
                                const float* __restrict__ g,
                                unsigned short* __restrict__ u, int n) {
  int i = blockIdx.x * 256 + threadIdx.x;
  if (i < n) {
    float av = a[i];
    float s = av / (1.f + __expf(-av));
    u[i] = f2bf(s * g[i]);
  }
}

// ---------------- x_out[n] = sum_m res[n,m]*x[m] + post[n]*y ----------------
__global__ __launch_bounds__(256) void combine_kernel(
    const float* __restrict__ xin, const float* __restrict__ res,
    const float* __restrict__ post, const float* __restrict__ y,
    float* __restrict__ xout)
{
  const int t = blockIdx.x, tid = threadIdx.x;
  __shared__ float rs[16], ps[4];
  if (tid < 16) rs[tid] = res[(size_t)t * 16 + tid];
  if (tid < 4)  ps[tid] = post[(size_t)t * 4 + tid];
  __syncthreads();
#pragma unroll
  for (int ci = 0; ci < 4; ci++) {
    int c = tid + ci * 256;
    float yv = y[(size_t)t * CC + c];
    float x0 = xin[(size_t)t * NC + 0*CC + c];
    float x1 = xin[(size_t)t * NC + 1*CC + c];
    float x2 = xin[(size_t)t * NC + 2*CC + c];
    float x3 = xin[(size_t)t * NC + 3*CC + c];
#pragma unroll
    for (int n = 0; n < 4; n++) {
      xout[(size_t)t * NC + n*CC + c] =
          rs[n*4+0]*x0 + rs[n*4+1]*x1 + rs[n*4+2]*x2 + rs[n*4+3]*x3 + ps[n]*yv;
    }
  }
}

// ---------------- host orchestration ----------------
extern "C" void kernel_launch(void* const* d_in, const int* in_sizes, int n_in,
                              void* d_out, int out_size, void* d_ws, size_t ws_size,
                              hipStream_t stream) {
  (void)in_sizes; (void)n_in; (void)out_size; (void)ws_size;

  const float* x    = (const float*)d_in[0];
  const int*   pos  = (const int*)d_in[1];
  const float* WQ   = (const float*)d_in[2];
  const float* WK   = (const float*)d_in[3];
  const float* WV   = (const float*)d_in[4];
  const float* WO   = (const float*)d_in[5];
  const float* ln1  = (const float*)d_in[6];
  const float* ln2  = (const float*)d_in[7];
  const float* W1   = (const float*)d_in[8];
  const float* W2   = (const float*)d_in[9];
  const float* W3   = (const float*)d_in[10];
  // mhc_attn: 11..19, mhc_ffn: 20..28 (dict insertion order)
  const float* Aphi_pre = (const float*)d_in[11];
  const float* Aphi_post= (const float*)d_in[12];
  const float* Aphi_res = (const float*)d_in[13];
  const float* Aa_pre   = (const float*)d_in[14];
  const float* Aa_post  = (const float*)d_in[15];
  const float* Aa_res   = (const float*)d_in[16];
  const float* Ab_pre   = (const float*)d_in[17];
  const float* Ab_post  = (const float*)d_in[18];
  const float* Ab_res   = (const float*)d_in[19];
  const float* Fphi_pre = (const float*)d_in[20];
  const float* Fphi_post= (const float*)d_in[21];
  const float* Fphi_res = (const float*)d_in[22];
  const float* Fa_pre   = (const float*)d_in[23];
  const float* Fa_post  = (const float*)d_in[24];
  const float* Fa_res   = (const float*)d_in[25];
  const float* Fb_pre   = (const float*)d_in[26];
  const float* Fb_post  = (const float*)d_in[27];
  const float* Fb_res   = (const float*)d_in[28];

  char* w = (char*)d_ws;
  size_t off = 0;
  auto alloc = [&](size_t bytes) {
    size_t r = off;
    off = (off + bytes + 255) & ~(size_t)255;
    return r;
  };

  const size_t szWcc  = (size_t)CC * CC * 2;     // bf16 C x C weight
  const size_t szWfcP = (size_t)FPAD * CC * 2;   // bf16 Fpad x C weight (padded)
  const size_t szWcf  = (size_t)CC * FF * 2;     // bf16 C x F weight (N=1024, no pad)
  const size_t szVec4 = (size_t)TT * 4 * 4;
  const size_t szRes  = (size_t)TT * 16 * 4;
  const size_t szHn   = (size_t)TT * CC * 2;
  const size_t szX    = (size_t)TT * NC * 4;
  const size_t szTCf  = (size_t)TT * CC * 4;
  const size_t szTCb  = (size_t)TT * CC * 2;
  const size_t szTFf  = (size_t)TT * FF * 4;
  const size_t szTFb  = (size_t)TT * FF * 2;

  size_t wqb = alloc(szWcc), wkb = alloc(szWcc), wvb = alloc(szWcc), wob = alloc(szWcc);
  size_t w1b = alloc(szWfcP), w2b = alloc(szWcf), w3b = alloc(szWfcP);
  size_t pre1 = alloc(szVec4), post1 = alloc(szVec4), res1 = alloc(szRes);
  size_t pre2 = alloc(szVec4), post2 = alloc(szVec4), res2 = alloc(szRes);
  size_t hnb = alloc(szHn);
  size_t x2o = alloc(szX);
  size_t scratch0 = off;
  // attention-phase scratch
  size_t qfo = alloc(szTCf), kfo = alloc(szTCf), vfo = alloc(szTCf);
  size_t qbo = alloc(szTCb), kbo = alloc(szTCb), vto = alloc(szTCb);
  size_t aoo = alloc(szTCb);
  size_t yfo = alloc(szTCf);
  // ffn-phase scratch (aliases attention scratch — attention data is dead)
  off = scratch0;
  size_t afo = alloc(szTFf), gfo = alloc(szTFf);
  size_t ubo = alloc(szTFb);
  size_t y2o = alloc(szTCf);

  auto US = [&](size_t o) { return (unsigned short*)(w + o); };
  auto FP = [&](size_t o) { return (float*)(w + o); };

  // ---- weight conversion to bf16 (padded rows zero-filled) ----
  {
    int ncc = CC * CC, nfc = FF * CC, nfcp = FPAD * CC;
    int gcc = (ncc + 255) / 256, gfcp = (nfcp + 255) / 256, gcf = (nfc + 255) / 256;
    cvt_bf16_kernel<<<gcc, 256, 0, stream>>>(WQ, US(wqb), ncc, ncc);
    cvt_bf16_kernel<<<gcc, 256, 0, stream>>>(WK, US(wkb), ncc, ncc);
    cvt_bf16_kernel<<<gcc, 256, 0, stream>>>(WV, US(wvb), ncc, ncc);
    cvt_bf16_kernel<<<gcc, 256, 0, stream>>>(WO, US(wob), ncc, ncc);
    cvt_bf16_kernel<<<gfcp, 256, 0, stream>>>(W1, US(w1b), nfc, nfcp);
    cvt_bf16_kernel<<<gcf, 256, 0, stream>>>(W2, US(w2b), nfc, nfc);
    cvt_bf16_kernel<<<gfcp, 256, 0, stream>>>(W3, US(w3b), nfc, nfcp);
  }

  // ---- stage 1: attention mHC ----
  mhc_stage_kernel<<<TT, 256, 0, stream>>>(
      x, Aphi_pre, Aphi_post, Aphi_res, Aa_pre, Aa_post, Aa_res,
      Ab_pre, Ab_post, Ab_res, ln1,
      FP(pre1), FP(post1), FP(res1), US(hnb));

  dim3 gQ((CC + 127) / 128, TT / 128);
  gemm_bf16_kernel<<<gQ, 256, 0, stream>>>(US(hnb), US(wqb), FP(qfo), TT, CC, CC);
  gemm_bf16_kernel<<<gQ, 256, 0, stream>>>(US(hnb), US(wkb), FP(kfo), TT, CC, CC);
  gemm_bf16_kernel<<<gQ, 256, 0, stream>>>(US(hnb), US(wvb), FP(vfo), TT, CC, CC);

  rope_reformat_kernel<<<(TT * HH + 255) / 256, 256, 0, stream>>>(
      FP(qfo), FP(kfo), FP(vfo), pos, US(qbo), US(kbo), US(vto));

  dim3 gA(BB * HH, LL / 64);
  flash_attn_kernel<<<gA, 128, 0, stream>>>(US(qbo), US(kbo), US(vto), US(aoo));

  gemm_bf16_kernel<<<gQ, 256, 0, stream>>>(US(aoo), US(wob), FP(yfo), TT, CC, CC);

  combine_kernel<<<TT, 256, 0, stream>>>(x, FP(res1), FP(post1), FP(yfo), FP(x2o));

  // ---- stage 2: FFN mHC ----
  mhc_stage_kernel<<<TT, 256, 0, stream>>>(
      FP(x2o), Fphi_pre, Fphi_post, Fphi_res, Fa_pre, Fa_post, Fa_res,
      Fb_pre, Fb_post, Fb_res, ln2,
      FP(pre2), FP(post2), FP(res2), US(hnb));

  dim3 gF((FF + 127) / 128, TT / 128);
  gemm_bf16_kernel<<<gF, 256, 0, stream>>>(US(hnb), US(w1b), FP(afo), TT, FF, CC);
  gemm_bf16_kernel<<<gF, 256, 0, stream>>>(US(hnb), US(w3b), FP(gfo), TT, FF, CC);

  int nTF = TT * FF;
  silu_mul_kernel<<<(nTF + 255) / 256, 256, 0, stream>>>(FP(afo), FP(gfo), US(ubo), nTF);

  gemm_bf16_kernel<<<gQ, 256, 0, stream>>>(US(ubo), US(w2b), FP(y2o), TT, CC, FF);

  combine_kernel<<<TT, 256, 0, stream>>>(FP(x2o), FP(res2), FP(post2), FP(y2o),
                                         (float*)d_out);
}
